// DySnakeConv_36464272343603
// MI455X (gfx1250) — matrix operand order repaired and checked by reading, not compiled, not run on hardware
//
#include <hip/hip_runtime.h>
#include <math.h>

#define EPS 1e-5f

typedef __attribute__((ext_vector_type(16))) __bf16        v16bf;
typedef __attribute__((ext_vector_type(8)))  float         v8f;
typedef __attribute__((ext_vector_type(4)))  unsigned int  v4u;

union BfPack { v4u u[2]; v16bf v; };

static constexpr int    S_  = 256;
static constexpr int    SS  = S_ * S_;

// ---- workspace layout (bytes), all offsets 256-aligned ----
static constexpr size_t WS_WMAIN = 0;                                  // 9*64*64 bf16
static constexpr size_t WS_WX2   = 73728;                              // 64*192 bf16
static constexpr size_t WS_WY2   = 98304;                              // 64*192 bf16
static constexpr size_t WS_SC0   = 122880;                             // 64 f32
static constexpr size_t WS_SH0   = 123136;                             // 64 f32
static constexpr size_t WS_OFFX  = 123392;                             // 4*3*SS f32
static constexpr size_t WS_OFFY  = WS_OFFX + (size_t)4*3*SS*4;
static constexpr size_t WS_Z     = WS_OFFY + (size_t)4*3*SS*4;         // 2*4*64*SS f32
static constexpr size_t WS_STATS = WS_Z + (size_t)2*4*64*SS*4;         // 128*2 f32

static __device__ __forceinline__ unsigned short f2bf(float f) {
    unsigned int u = __float_as_uint(f);
    u = (u + 0x7FFFu + ((u >> 16) & 1u)) >> 16;   // round-to-nearest-even
    return (unsigned short)u;
}

// ---------------- weight repack + BN fold ----------------
__global__ void pack_kernel(const float* __restrict__ w0, const float* __restrict__ g0,
                            const float* __restrict__ b0, const float* __restrict__ m0,
                            const float* __restrict__ v0, const float* __restrict__ wx,
                            const float* __restrict__ wy, char* ws) {
    unsigned short* Wm  = (unsigned short*)(ws + WS_WMAIN);
    unsigned short* Wx2 = (unsigned short*)(ws + WS_WX2);
    unsigned short* Wy2 = (unsigned short*)(ws + WS_WY2);
    float* sc = (float*)(ws + WS_SC0);
    float* sh = (float*)(ws + WS_SH0);
    int tid = blockIdx.x * blockDim.x + threadIdx.x;
    int stride = gridDim.x * blockDim.x;
    for (int e = tid; e < 9 * 64 * 64; e += stride) {            // Wmain[tap][o][c]
        int c = e % 64, o = (e / 64) % 64, t = e / 4096;
        Wm[(t * 64 + o) * 64 + c] = f2bf(w0[(o * 64 + c) * 9 + t]);
    }
    for (int e = tid; e < 64 * 192; e += stride) {               // W?2[o][p*64+c]
        int c = e % 64, p = (e / 64) % 3, o = e / 192;
        Wx2[o * 192 + p * 64 + c] = f2bf(wx[(o * 64 + c) * 3 + p]);
        Wy2[o * 192 + p * 64 + c] = f2bf(wy[(o * 64 + c) * 3 + p]);
    }
    for (int o = tid; o < 64; o += stride) {
        float s = g0[o] * rsqrtf(v0[o] + EPS);
        sc[o] = s;
        sh[o] = b0[o] - m0[o] * s;
    }
}

// ---------------- offset field: conv3x3(64->3 per branch) + folded BN + tanh ----------------
__global__ __launch_bounds__(256)
void offset_kernel(const float* __restrict__ x,
                   const float* owx, const float* obx, const float* ogx,
                   const float* obbx, const float* omx, const float* ovx,
                   const float* owy, const float* oby, const float* ogy,
                   const float* obby, const float* omy, const float* ovy,
                   char* ws) {
    __shared__ float wX[3][64][9];
    __shared__ float wY[3][64][9];
    __shared__ float fsX[3], fbX[3], fsY[3], fbY[3];
    int b = blockIdx.x / 256, i = blockIdx.x % 256;
    int j = threadIdx.x;
    for (int e = threadIdx.x; e < 3 * 64 * 9; e += 256) {
        int kk = e % 9, c = (e / 9) % 64, p = e / 576;
        wX[p][c][kk] = owx[(p * 64 + c) * 9 + kk];          // morph0 uses channels 0..2 of x-branch
        wY[p][c][kk] = owy[((3 + p) * 64 + c) * 9 + kk];    // morph1 uses channels 3..5 of y-branch
    }
    if (threadIdx.x < 3) {
        int p = threadIdx.x;
        float s = ogx[p] * rsqrtf(ovx[p] + EPS);
        fsX[p] = s; fbX[p] = (obx[p] - omx[p]) * s + obbx[p];
        int q = 3 + p;
        s = ogy[q] * rsqrtf(ovy[q] + EPS);
        fsY[p] = s; fbY[p] = (oby[q] - omy[q]) * s + obby[q];
    }
    __syncthreads();
    float accX[3] = {0.f, 0.f, 0.f}, accY[3] = {0.f, 0.f, 0.f};
    const float* xb = x + (size_t)b * 64 * SS;
    for (int c = 0; c < 64; ++c) {
        float v[9];
#pragma unroll
        for (int r = 0; r < 3; ++r)
#pragma unroll
            for (int q = 0; q < 3; ++q) {
                int y = i - 1 + r, xx = j - 1 + q;
                v[r * 3 + q] = (y >= 0 && y < 256 && xx >= 0 && xx < 256)
                                   ? xb[c * SS + y * 256 + xx] : 0.f;
            }
#pragma unroll
        for (int p = 0; p < 3; ++p) {
            float ax = 0.f, ay = 0.f;
#pragma unroll
            for (int kk = 0; kk < 9; ++kk) { ax += wX[p][c][kk] * v[kk]; ay += wY[p][c][kk] * v[kk]; }
            accX[p] += ax; accY[p] += ay;
        }
    }
    float* OX = (float*)(ws + WS_OFFX);
    float* OY = (float*)(ws + WS_OFFY);
#pragma unroll
    for (int p = 0; p < 3; ++p) {
        // _apply_offset zeroes the center tap (cumulative loop empty for k=3), extend_scope=1
        float vx2 = (p == 1) ? 0.f : tanhf(accX[p] * fsX[p] + fbX[p]);
        float vy2 = (p == 1) ? 0.f : tanhf(accY[p] * fsY[p] + fbY[p]);
        size_t o = (((size_t)b * 3 + p) * 256 + i) * 256 + j;
        OX[o] = vx2; OY[o] = vy2;
    }
}

// ---------------- main 3x3 conv as 9-tap implicit GEMM, bf16 WMMA, fused BN+SiLU ----------------
__global__ __launch_bounds__(256)
void conv_main_kernel(const float* __restrict__ x, const char* __restrict__ ws,
                      float* __restrict__ out) {
    __shared__ __align__(16) unsigned short patch[3 * 34 * 64];   // [row][col][channel] bf16
    int blk = blockIdx.x;
    int wt = blk % 8;
    int i  = (blk / 8) % 256;
    int b  = blk / 2048;
    int wbase = wt * 32;
    const float* xb = x + (size_t)b * 64 * SS;
    for (int e = threadIdx.x; e < 3 * 34 * 64; e += 256) {
        int col = e % 34, r = (e / 34) % 3, c = e / 102;
        int y = i - 1 + r, xx = wbase - 1 + col;
        float v = (y >= 0 && y < 256 && xx >= 0 && xx < 256) ? xb[c * SS + y * 256 + xx] : 0.f;
        patch[(r * 34 + col) * 64 + c] = f2bf(v);
    }
    __syncthreads();
    int wave = threadIdx.x >> 5, lane = threadIdx.x & 31;
    int mt = wave & 3, nt = wave >> 2;
    int Mbase = mt * 16;
    int nloc  = nt * 16 + (lane & 15);
    int hi    = lane >> 4;
    const unsigned short* Wm = (const unsigned short*)(ws + WS_WMAIN);
    v8f acc = {0.f, 0.f, 0.f, 0.f, 0.f, 0.f, 0.f, 0.f};
#pragma unroll
    for (int r = 0; r < 3; ++r)
#pragma unroll
        for (int q = 0; q < 3; ++q) {
            int tap = r * 3 + q;
            const unsigned short* bbase = &patch[(r * 34 + nloc + q) * 64];
            const unsigned short* abase = &Wm[(tap * 64 + Mbase + (lane & 15)) * 64];
#pragma unroll
            for (int s = 0; s < 2; ++s) {
                BfPack A, Bm;
                // A 16x32 bf16: lane-half holds K {hi*8..+7, 16+hi*8..+7} of row M
                A.u[0] = *(const v4u*)(abase + s * 32 + hi * 8);
                A.u[1] = *(const v4u*)(abase + s * 32 + 16 + hi * 8);
                // B 32x16 bf16: lane-half holds K {hi*8..+7, 16+hi*8..+7} of column N
                Bm.u[0] = *(const v4u*)(bbase + s * 32 + hi * 8);
                Bm.u[1] = *(const v4u*)(bbase + s * 32 + 16 + hi * 8);
                acc = __builtin_amdgcn_wmma_f32_16x16x32_bf16(false, A.v, false, Bm.v,
                                                              (short)0, acc, false, false);
            }
        }
    const float* sc = (const float*)(ws + WS_SC0);
    const float* sh = (const float*)(ws + WS_SH0);
    int jcol = wbase + nloc;
#pragma unroll
    for (int vr = 0; vr < 8; ++vr) {
        int o = Mbase + vr + hi * 8;                  // C/D layout: VGPR r -> M=r (lanes 0-15) / 8+r
        float y = acc[vr] * sc[o] + sh[o];
        out[(((size_t)b * 192 + o) * 256 + i) * 256 + jcol] = y / (1.f + __expf(-y));
    }
}

// ---------------- snake branch: bilinear gather (from batch 0) + K=192 GEMM ----------------
__global__ __launch_bounds__(256)
void dsconv_kernel(const float* __restrict__ x, char* ws,
                   const float* __restrict__ bw, int morph) {
    __shared__ __align__(16) unsigned short G[32][192];           // [pixel][p*64+c] bf16
    int blk = blockIdx.x;
    int jt = blk % 8;
    int i  = (blk / 8) % 256;
    int b  = blk / 2048;
    int jbase = jt * 32;
    const float* OFF = (const float*)(ws + (morph == 0 ? WS_OFFX : WS_OFFY));
    const float vals[3] = {-2.f, -0.5f, 1.f};                     // source-quirk linspace(-2,1,3)
    int n = threadIdx.x >> 3;
    int oct = threadIdx.x & 7;
    int j = jbase + n;
#pragma unroll
    for (int p = 0; p < 3; ++p) {
        float off = OFF[(((size_t)b * 3 + p) * 256 + i) * 256 + j];
        float yf, xf;
        if (morph == 0) { yf = (float)i + off;            xf = (float)i + vals[p]; }
        else            { yf = (float)i + vals[p] + off;  xf = (float)i; }
        float fy = floorf(yf), fx = floorf(xf);
        float y0 = fminf(fmaxf(fy, 0.f), 255.f);
        float y1 = fminf(fmaxf(fy + 1.f, 0.f), 255.f);
        float x0 = fminf(fmaxf(fx, 0.f), 255.f);
        float x1 = fminf(fmaxf(fx + 1.f, 0.f), 255.f);
        float w00 = (y1 - yf) * (x1 - xf), w01 = (y1 - yf) * (xf - x0);
        float w10 = (yf - y0) * (x1 - xf), w11 = (yf - y0) * (xf - x0);
        int a00 = (int)y0 * 256 + (int)x0, a01 = (int)y0 * 256 + (int)x1;
        int a10 = (int)y1 * 256 + (int)x0, a11 = (int)y1 * 256 + (int)x1;
#pragma unroll
        for (int cc = 0; cc < 8; ++cc) {
            int c = oct * 8 + cc;
            const float* xc = x + (size_t)c * SS;                 // batch 0 (source quirk)
            float v = w00 * xc[a00] + w01 * xc[a01] + w10 * xc[a10] + w11 * xc[a11];
            G[n][p * 64 + c] = f2bf(v);
        }
    }
    __syncthreads();
    int wave = threadIdx.x >> 5, lane = threadIdx.x & 31;
    int mt = wave & 3, nt = wave >> 2;
    int Mbase = mt * 16;
    int nloc  = nt * 16 + (lane & 15);
    int hi    = lane >> 4;
    const unsigned short* W2 = (const unsigned short*)(ws + (morph == 0 ? WS_WX2 : WS_WY2));
    const unsigned short* arow = &W2[(Mbase + (lane & 15)) * 192];
    const unsigned short* brow = &G[nloc][0];
    v8f acc = {0.f, 0.f, 0.f, 0.f, 0.f, 0.f, 0.f, 0.f};
#pragma unroll
    for (int s = 0; s < 6; ++s) {
        BfPack A, Bm;
        A.u[0] = *(const v4u*)(arow + s * 32 + hi * 8);
        A.u[1] = *(const v4u*)(arow + s * 32 + 16 + hi * 8);
        Bm.u[0] = *(const v4u*)(brow + s * 32 + hi * 8);
        Bm.u[1] = *(const v4u*)(brow + s * 32 + 16 + hi * 8);
        acc = __builtin_amdgcn_wmma_f32_16x16x32_bf16(false, A.v, false, Bm.v,
                                                      (short)0, acc, false, false);
    }
    float* Z = (float*)(ws + WS_Z) + ((size_t)morph * 4 + b) * 64 * SS;
    int jcol = jbase + nloc;
#pragma unroll
    for (int vr = 0; vr < 8; ++vr) {
        int o = Mbase + vr + hi * 8;
        Z[(size_t)o * SS + i * 256 + jcol] = acc[vr] + bw[o];
    }
}

// ---------------- GroupNorm statistics (groups of 4 channels) ----------------
__global__ __launch_bounds__(256)
void gn_stats_kernel(char* ws) {
    int g  = blockIdx.x % 16;
    int mb = blockIdx.x / 16;                                     // morph*4 + b
    const float* Z = (const float*)(ws + WS_Z) + ((size_t)mb * 64 + g * 4) * SS;
    float s = 0.f, ss = 0.f;
    for (int e = threadIdx.x; e < 4 * SS; e += 256) {
        float v = Z[e];
        s += v; ss += v * v;
    }
    __shared__ float rs[256], rss[256];
    rs[threadIdx.x] = s; rss[threadIdx.x] = ss;
    __syncthreads();
    for (int off = 128; off > 0; off >>= 1) {
        if (threadIdx.x < off) {
            rs[threadIdx.x]  += rs[threadIdx.x + off];
            rss[threadIdx.x] += rss[threadIdx.x + off];
        }
        __syncthreads();
    }
    if (threadIdx.x == 0) {
        float inv = 1.f / (4.f * SS);
        float mu  = rs[0] * inv;
        float var = rss[0] * inv - mu * mu;
        float* st = (float*)(ws + WS_STATS) + (size_t)blockIdx.x * 2;
        st[0] = mu; st[1] = rsqrtf(var + EPS);
    }
}

// ---------------- GroupNorm apply + SiLU into concat output ----------------
__global__ __launch_bounds__(256)
void gn_finalize_kernel(char* ws, const float* __restrict__ ggx, const float* __restrict__ gbx,
                        const float* __restrict__ ggy, const float* __restrict__ gby,
                        float* __restrict__ out) {
    size_t idx = (size_t)blockIdx.x * 256 + threadIdx.x;          // over 2*4*64*SS
    int pix   = (int)(idx % SS);
    int o     = (int)((idx / SS) % 64);
    int b     = (int)((idx / ((size_t)SS * 64)) % 4);
    int morph = (int)(idx / ((size_t)SS * 64 * 4));
    float z = ((const float*)(ws + WS_Z))[idx];
    const float* st = (const float*)(ws + WS_STATS) + (((size_t)morph * 4 + b) * 16 + o / 4) * 2;
    const float* gg = morph ? ggy : ggx;
    const float* gb = morph ? gby : gbx;
    float y = (z - st[0]) * st[1] * gg[o] + gb[o];
    out[(((size_t)b * 192 + 64 + morph * 64 + o) * SS) + pix] = y / (1.f + __expf(-y));
}

extern "C" void kernel_launch(void* const* d_in, const int* in_sizes, int n_in,
                              void* d_out, int out_size, void* d_ws, size_t ws_size,
                              hipStream_t stream) {
    const float* x    = (const float*)d_in[0];
    const float* w0   = (const float*)d_in[1];
    const float* g0   = (const float*)d_in[2];
    const float* b0   = (const float*)d_in[3];
    const float* m0   = (const float*)d_in[4];
    const float* v0   = (const float*)d_in[5];
    const float* owx  = (const float*)d_in[6];
    const float* obx  = (const float*)d_in[7];
    const float* ogx  = (const float*)d_in[8];
    const float* obbx = (const float*)d_in[9];
    const float* omx  = (const float*)d_in[10];
    const float* ovx  = (const float*)d_in[11];
    const float* wx   = (const float*)d_in[12];
    const float* bwx  = (const float*)d_in[13];
    const float* ggx  = (const float*)d_in[14];
    const float* gbx  = (const float*)d_in[15];
    const float* owy  = (const float*)d_in[16];
    const float* oby  = (const float*)d_in[17];
    const float* ogy  = (const float*)d_in[18];
    const float* obby = (const float*)d_in[19];
    const float* omy  = (const float*)d_in[20];
    const float* ovy  = (const float*)d_in[21];
    const float* wy   = (const float*)d_in[22];
    const float* bwy  = (const float*)d_in[23];
    const float* ggy  = (const float*)d_in[24];
    const float* gby  = (const float*)d_in[25];
    char* ws = (char*)d_ws;
    float* out = (float*)d_out;

    pack_kernel<<<64, 256, 0, stream>>>(w0, g0, b0, m0, v0, wx, wy, ws);
    offset_kernel<<<1024, 256, 0, stream>>>(x, owx, obx, ogx, obbx, omx, ovx,
                                            owy, oby, ogy, obby, omy, ovy, ws);
    conv_main_kernel<<<8192, 256, 0, stream>>>(x, ws, out);
    dsconv_kernel<<<8192, 256, 0, stream>>>(x, ws, bwx, 0);
    dsconv_kernel<<<8192, 256, 0, stream>>>(x, ws, bwy, 1);
    gn_stats_kernel<<<128, 256, 0, stream>>>(ws);
    gn_finalize_kernel<<<131072, 256, 0, stream>>>(ws, ggx, gbx, ggy, gby, out);
}
